// CrossmodalModule_43911745635137
// MI455X (gfx1250) — matrix operand order repaired and checked by reading, not compiled, not run on hardware
//
#include <hip/hip_runtime.h>
#include <hip/hip_bf16.h>

// ---------------------------------------------------------------------------
// Problem constants (from the reference)
// ---------------------------------------------------------------------------
#define BB      16
#define LLANG   8
#define PPROP   256
#define TTOK    64
#define HH      128
#define NHEADS  4
#define DKH     32            // HH / NHEADS
#define BL      (BB * LLANG)  // 128
#define RA      (BL * PPROP)  // 32768 rows of A
#define RB      (BL * TTOK)   // 8192  rows of Bf
#define SCALE_QK 0.17677669529663687f  // 1/sqrt(32)

typedef __attribute__((ext_vector_type(16))) _Float16 v16h;
typedef __attribute__((ext_vector_type(2)))  _Float16 h2;
typedef __attribute__((ext_vector_type(8)))  float    v8f;

// ---------------------------------------------------------------------------
// WMMA fragment loaders (CDNA5 ISA 7.12.2 layouts, wave32)
// ---------------------------------------------------------------------------
// A fragment 16x32 f16 from row-major src (row stride ld, K contiguous)
__device__ __forceinline__ v16h frag_a_h(const _Float16* __restrict__ src, int ld) {
  const int l = threadIdx.x & 31;
  const int m = l & 15, half = l >> 4;
  const h2* p = (const h2*)(src + (size_t)m * ld);
  v16h out;
#pragma unroll
  for (int v = 0; v < 8; ++v) {
    const int k = 2 * v + 16 * half;  // contiguous k = 16*half + s (matches B)
    h2 t = p[k >> 1];
    out[2 * v] = t[0]; out[2 * v + 1] = t[1];
  }
  return out;
}

// B fragment 32x16: element (k,n) = src[n*ld + k]  (k contiguous; K^T / Vt case)
// Per lane this is 16 contiguous halves -> 32B vector load.
__device__ __forceinline__ v16h frag_b_t(const _Float16* __restrict__ src, int ld) {
  const int l = threadIdx.x & 31;
  const int n = l & 15, half = l >> 4;
  const h2* p = (const h2*)(src + (size_t)n * ld + 16 * half);
  v16h out;
#pragma unroll
  for (int v = 0; v < 8; ++v) { h2 t = p[v]; out[2 * v] = t[0]; out[2 * v + 1] = t[1]; }
  return out;
}

__device__ __forceinline__ v8f wmma_f16(v16h a, v16h b, v8f c) {
  return __builtin_amdgcn_wmma_f32_16x16x32_f16(false, a, false, b, (short)0, c, false, false);
}

// ---------------------------------------------------------------------------
// Workgroup WMMA GEMM: Y(R,128) = X(R,128) @ W + bias
//   * W pre-packed into fragment-native layout, staged in LDS per block
//   * block = 256 threads = 8 waves; each wave computes a 16 x (NT*16) strip
//   * all NT B-fragments issued before the WMMA chain (overlap DS latency)
//   * template booleans -> branch-free store path
//   * TRANS: store f16 output channel-major per batch (for V -> Vt)
// ---------------------------------------------------------------------------
template<int NT, bool F32OUT, bool F16OUT, bool TRANS, bool PARTIAL>
__global__ __launch_bounds__(256) void gemm_ws(const _Float16* __restrict__ X,
                                               const _Float16* __restrict__ Wp,
                                               const float* __restrict__ bias,
                                               float* __restrict__ Yf,
                                               _Float16* __restrict__ Yh,
                                               int Nout, int ldY, int nkb) {
  __shared__ v16h wlds[NT * 4 * 32];  // NT*4 frags * 32 lanes (32B each)
  {
    const uint4* s = (const uint4*)Wp;
    uint4* d = (uint4*)wlds;
    const int tot = NT * 4 * 32 * 2;  // uint4 count
#pragma unroll 2
    for (int i = threadIdx.x; i < tot; i += 256) d[i] = s[i];
  }
  __syncthreads();

  const int wave = threadIdx.x >> 5;
  const int lane = threadIdx.x & 31;
  const int r0 = (blockIdx.x * 8 + wave) * 16;

  v8f acc[NT];
  const v8f vzero = {0.f, 0.f, 0.f, 0.f, 0.f, 0.f, 0.f, 0.f};
#pragma unroll
  for (int t = 0; t < NT; ++t) acc[t] = vzero;

#pragma unroll
  for (int ks = 0; ks < 4; ++ks) {              // K = 128 = 4 * 32
    v16h a = frag_a_h(X + (size_t)r0 * HH + ks * 32, HH);
    v16h b[NT];
#pragma unroll
    for (int t = 0; t < NT; ++t) b[t] = wlds[(t * 4 + ks) * 32 + lane];  // issue all DS loads
#pragma unroll
    for (int t = 0; t < NT; ++t) acc[t] = wmma_f16(a, b[t], acc[t]);     // then compute
  }

  const int n_lo = lane & 15, half = lane >> 4;
#pragma unroll
  for (int t = 0; t < NT; ++t) {
    const int n = t * 16 + n_lo;
    if (PARTIAL && n >= Nout) continue;
    const float bv = bias[n];
#pragma unroll
    for (int v = 0; v < 8; ++v) {
      const int row = r0 + v + 8 * half;
      const float val = acc[t][v] + bv;
      if (F32OUT) Yf[(size_t)row * ldY + n] = val;
      if (F16OUT) {
        if (TRANS) {
          const int bl = row / nkb, kk = row - bl * nkb;
          Yh[((size_t)bl * HH + n) * nkb + kk] = (_Float16)val;  // Vt[bl][c][k]
        } else {
          Yh[(size_t)row * ldY + n] = (_Float16)val;
        }
      }
    }
  }
}

// ---------------------------------------------------------------------------
// Fused attention: scores (WMMA) -> LDS, softmax in LDS, probs @ Vt (WMMA).
// One block = one (batch-head, 16-query tile). block = 256 threads = 8 waves.
// grid = (nq/16, BL*NHEADS). NK = key count (64 or 256).
// ---------------------------------------------------------------------------
template<int NK>
__global__ __launch_bounds__(256) void attn_fused(const _Float16* __restrict__ Qh,
                                                  const _Float16* __restrict__ Kh,
                                                  const _Float16* __restrict__ Vt,
                                                  const int* __restrict__ mask,
                                                  _Float16* __restrict__ Oh, int nq) {
  __shared__ float sc[16][NK];                    // f32 scores
  __shared__ __align__(16) _Float16 pr[16][NK];   // f16 probabilities
  __shared__ float red[16][17];                   // row reduction scratch (padded)
  __shared__ float rowstat[16];

  const int bh = blockIdx.y, bl = bh >> 2, h = bh & 3;
  const int q0 = blockIdx.x * 16;
  const int wave = threadIdx.x >> 5, lane = threadIdx.x & 31;
  const int n_lo = lane & 15, half = lane >> 4;
  const v8f vzero = {0.f, 0.f, 0.f, 0.f, 0.f, 0.f, 0.f, 0.f};

  // ---- Phase 1: score tiles -> LDS (k-tiles distributed over waves) ----
  v16h qa = frag_a_h(Qh + ((size_t)(bl * nq + q0)) * HH + h * DKH, HH);
  for (int kt = wave; kt < NK / 16; kt += 8) {
    const int k0 = kt * 16;
    v16h kb = frag_b_t(Kh + ((size_t)(bl * NK + k0)) * HH + h * DKH, HH);
    v8f c = vzero;
    c = wmma_f16(qa, kb, c);
    const int kc = k0 + n_lo;
    const bool msk = (mask != nullptr) && (mask[bl * NK + kc] != 0);
#pragma unroll
    for (int v = 0; v < 8; ++v)
      sc[v + 8 * half][kc] = msk ? -1e9f : c[v] * SCALE_QK;
  }
  __syncthreads();

  // ---- Phase 2: softmax per row (16 threads per row) ----
  const int r  = threadIdx.x >> 4;  // row 0..15
  const int tt = threadIdx.x & 15;  // slot within row
  constexpr int EPT = NK / 16;      // elements per thread
  float loc[EPT];
  float mx = -3.402823466e38f;
#pragma unroll
  for (int i = 0; i < EPT; ++i) { loc[i] = sc[r][tt + 16 * i]; mx = fmaxf(mx, loc[i]); }
  red[r][tt] = mx; __syncthreads();
  if (tt == 0) {
    float m2 = red[r][0];
#pragma unroll
    for (int i = 1; i < 16; ++i) m2 = fmaxf(m2, red[r][i]);
    rowstat[r] = m2;
  }
  __syncthreads();
  mx = rowstat[r];
  float sum = 0.f;
#pragma unroll
  for (int i = 0; i < EPT; ++i) { loc[i] = expf(loc[i] - mx); sum += loc[i]; }
  __syncthreads();  // all rowstat reads done before it is rewritten
  red[r][tt] = sum; __syncthreads();
  if (tt == 0) {
    float s2 = 0.f;
#pragma unroll
    for (int i = 0; i < 16; ++i) s2 += red[r][i];
    rowstat[r] = s2;
  }
  __syncthreads();
  const float inv = 1.f / rowstat[r];
#pragma unroll
  for (int i = 0; i < EPT; ++i) pr[r][tt + 16 * i] = (_Float16)(loc[i] * inv);
  __syncthreads();

  // ---- Phase 3: out(16 x 32) = probs @ Vt^T ; waves 0,1 take one d-tile each ----
  if (wave < 2) {
    const int d0 = wave * 16;
    const _Float16* vb = Vt + ((size_t)bl * HH + h * DKH + d0) * NK;
    v8f c = vzero;
#pragma unroll
    for (int k0 = 0; k0 < NK; k0 += 32) {
      v16h a = frag_a_h(&pr[0][0] + k0, NK);  // A-fragment from LDS
      v16h b = frag_b_t(vb + k0, NK);
      c = wmma_f16(a, b, c);
    }
#pragma unroll
    for (int v = 0; v < 8; ++v)
      Oh[((size_t)(bl * nq + q0 + v + 8 * half)) * HH + h * DKH + d0 + n_lo] = (_Float16)c[v];
  }
}

// ---------------------------------------------------------------------------
// Residual + LayerNorm over HH=128; block = 128 threads = 1 row. In-place X.
// ---------------------------------------------------------------------------
__global__ void residual_ln(const float* __restrict__ Y, float* __restrict__ X,
                            _Float16* __restrict__ Xh,
                            const float* __restrict__ g, const float* __restrict__ be) {
  __shared__ float sh[128];
  const size_t row = blockIdx.x;
  const int t = threadIdx.x;
  float x = X[row * HH + t] + Y[row * HH + t];
  sh[t] = x; __syncthreads();
  for (int s = 64; s > 0; s >>= 1) { if (t < s) sh[t] += sh[t + s]; __syncthreads(); }
  const float mu = sh[0] * (1.f / HH); __syncthreads();
  const float d = x - mu;
  sh[t] = d * d; __syncthreads();
  for (int s = 64; s > 0; s >>= 1) { if (t < s) sh[t] += sh[t + s]; __syncthreads(); }
  const float var = sh[0] * (1.f / HH);
  const float o = d * rsqrtf(var + 1e-5f) * g[t] + be[t];
  X[row * HH + t] = o;
  Xh[row * HH + t] = (_Float16)o;
}

// ---------------------------------------------------------------------------
// BatchNorm (batch statistics, biased var) + PReLU helpers
// ---------------------------------------------------------------------------
__global__ void bn_stats(const float* __restrict__ Y, int rows,
                         float* __restrict__ mean, float* __restrict__ var) {
  __shared__ float s1[256], s2[256];
  const int c = blockIdx.x, t = threadIdx.x;
  float a = 0.f, b = 0.f;
  for (int r = t; r < rows; r += 256) {
    const float v = Y[(size_t)r * HH + c];
    a += v; b += v * v;
  }
  s1[t] = a; s2[t] = b; __syncthreads();
  for (int s = 128; s > 0; s >>= 1) {
    if (t < s) { s1[t] += s1[t + s]; s2[t] += s2[t + s]; }
    __syncthreads();
  }
  if (t == 0) {
    const float mu = s1[0] / rows;
    mean[c] = mu;
    var[c] = s2[0] / rows - mu * mu;
  }
}

__global__ void bn_prelu(const float* __restrict__ Y, _Float16* __restrict__ Xh,
                         const float* __restrict__ mean, const float* __restrict__ var,
                         const float* __restrict__ g, const float* __restrict__ be,
                         const float* __restrict__ alpha, int total) {
  const int i = blockIdx.x * 256 + threadIdx.x;
  if (i >= total) return;
  const int c = i & (HH - 1);
  float y = (Y[i] - mean[c]) * rsqrtf(var[c] + 1e-5f) * g[c] + be[c];
  y = (y >= 0.f) ? y : alpha[0] * y;
  Xh[i] = (_Float16)y;
}

// ---------------------------------------------------------------------------
// Small data-movement kernels
// ---------------------------------------------------------------------------
__global__ void f32_to_h(const float* __restrict__ s, _Float16* __restrict__ d, int n) {
  const int i = blockIdx.x * 256 + threadIdx.x;
  if (i < n) d[i] = (_Float16)s[i];
}

__global__ void init_bf(const float* __restrict__ lang, float* __restrict__ Bf32,
                        _Float16* __restrict__ Bfh, int n) {
  const int i = blockIdx.x * 256 + threadIdx.x;
  if (i < n) { const float v = lang[i]; Bf32[i] = v; Bfh[i] = (_Float16)v; }
}

// replicate per-batch object features across L questions
__global__ void bcast_A(const float* __restrict__ A0, float* __restrict__ A32,
                        _Float16* __restrict__ Ah) {
  const int i = blockIdx.x * 256 + threadIdx.x;
  if (i >= RA * HH) return;
  const int ph = i % (PPROP * HH);
  const int bl = i / (PPROP * HH);
  const float v = A0[(size_t)(bl / LLANG) * (PPROP * HH) + ph];
  A32[i] = v; Ah[i] = (_Float16)v;
}

// Pack an f32 weight matrix (K=128 x Nreal, row-major) into B-fragment layout,
// zero-padding columns up to N (multiple of 16).
__global__ void pack_w(const float* __restrict__ w, _Float16* __restrict__ out,
                       int N, int Nreal) {
  const int total = (N / 16) * 4 * 512;
  const int i = blockIdx.x * 256 + threadIdx.x;
  if (i >= total) return;
  const int e = i & 15;
  const int l = (i >> 4) & 31;
  const int f = i >> 9;             // frag = nt*4 + ks
  const int ks = f & 3, nt = f >> 2;
  const int n = nt * 16 + (l & 15);
  const int k = ks * 32 + 16 * (l >> 4) + e;
  out[i] = (_Float16)((n < Nreal) ? w[k * Nreal + n] : 0.f);
}

// ---------------------------------------------------------------------------
// Host orchestration
// ---------------------------------------------------------------------------
extern "C" void kernel_launch(void* const* d_in, const int* in_sizes, int n_in,
                              void* d_out, int out_size, void* d_ws, size_t ws_size,
                              hipStream_t stream) {
  (void)in_sizes; (void)n_in; (void)out_size; (void)ws_size;

  // Input order: recursive dict-insertion-order flatten of setup_inputs().
  const float* bbox  = (const float*)d_in[0];
  const float* lang  = (const float*)d_in[1];
  const int*   qmask = (const int*)d_in[2];
  // d_in[3] = dist_weights -> dead code in the reference
  const float* objw  = (const float*)d_in[4];
  const float* objb  = (const float*)d_in[5];
  const float* c1w = (const float*)d_in[46]; const float* c1b = (const float*)d_in[47];
  const float* bn1g = (const float*)d_in[48]; const float* bn1b = (const float*)d_in[49];
  const float* pr1 = (const float*)d_in[50];
  const float* c2w = (const float*)d_in[51]; const float* c2b = (const float*)d_in[52];
  const float* bn2g = (const float*)d_in[53]; const float* bn2b = (const float*)d_in[54];
  const float* pr2 = (const float*)d_in[55];
  const float* c3w = (const float*)d_in[56]; const float* c3b = (const float*)d_in[57];

  // Workspace carve-up
  char* base = (char*)d_ws;
  size_t off = 0;
  auto take = [&](size_t bytes) -> void* {
    void* p = base + off;
    off = (off + bytes + 255) & ~(size_t)255;
    return p;
  };
  const int WMAT = HH * HH;  // packed size of a 128x128 matrix (halves)
  _Float16* WH    = (_Float16*)take((size_t)19 * WMAT * sizeof(_Float16)); // packed weights
  _Float16* W3H   = (_Float16*)take((size_t)4 * 512 * sizeof(_Float16));   // packed conv3 (N=16)
  _Float16* bboxh = (_Float16*)take((size_t)BB * PPROP * HH * sizeof(_Float16));
  float*    A32   = (float*)take((size_t)RA * HH * sizeof(float));
  _Float16* Ah    = (_Float16*)take((size_t)RA * HH * sizeof(_Float16));
  _Float16* Bfh   = (_Float16*)take((size_t)RB * HH * sizeof(_Float16));
  _Float16* Qh    = (_Float16*)take((size_t)RA * HH * sizeof(_Float16));
  _Float16* Kh    = (_Float16*)take((size_t)RA * HH * sizeof(_Float16));
  _Float16* Vt    = (_Float16*)take((size_t)RA * HH * sizeof(_Float16));  // channel-major V
  float*    Ytmp  = (float*)take((size_t)RA * HH * sizeof(float));
  float*    stats = (float*)take(256 * sizeof(float));

  // Output split: conf (BL*P*4) then Bf (BL*T*H). Bf's f32 state lives in d_out.
  float* conf = (float*)d_out;
  float* Bf32 = (float*)d_out + (size_t)BL * PPROP * 4;

  auto packw = [&](const float* s, _Float16* d, int N, int Nreal) {
    const int total = (N / 16) * 4 * 512;
    pack_w<<<(total + 255) / 256, 256, 0, stream>>>(s, d, N, Nreal);
  };

  // Weight slots in WH: 0=objfc, 1+4i+{0..3}={q,k,v,o} of MHA i, 17=conv1, 18=conv2
  packw(objw, WH, HH, HH);
  for (int i = 0; i < 4; ++i) {
    const int din = 6 + 10 * i;
    for (int j = 0; j < 4; ++j)
      packw((const float*)d_in[din + 2 * j], WH + (size_t)(1 + 4 * i + j) * WMAT, HH, HH);
  }
  packw(c1w, WH + (size_t)17 * WMAT, HH, HH);
  packw(c2w, WH + (size_t)18 * WMAT, HH, HH);
  packw(c3w, W3H, 16, 4);
  f32_to_h<<<(BB * PPROP * HH + 255) / 256, 256, 0, stream>>>(bbox, bboxh, BB * PPROP * HH);
  init_bf<<<(RB * HH + 255) / 256, 256, 0, stream>>>(lang, Bf32, Bfh, RB * HH);

  // A = object_fc(bbox), then broadcast over L
  gemm_ws<8, true, false, false, false><<<(BB * PPROP) / 128, 256, 0, stream>>>(
      bboxh, WH, objb, Ytmp, nullptr, HH, HH, 1);
  bcast_A<<<(RA * HH + 255) / 256, 256, 0, stream>>>(Ytmp, A32, Ah);

  auto run_mha = [&](int mi, float* Xf, _Float16* Xh, int nq,
                     const _Float16* KeysH, int nk, const int* mask) {
    const _Float16* Wq = WH + (size_t)(1 + 4 * mi + 0) * WMAT;
    const _Float16* Wk = WH + (size_t)(1 + 4 * mi + 1) * WMAT;
    const _Float16* Wv = WH + (size_t)(1 + 4 * mi + 2) * WMAT;
    const _Float16* Wo = WH + (size_t)(1 + 4 * mi + 3) * WMAT;
    const int din = 6 + 10 * mi;
    const float* qb  = (const float*)d_in[din + 1];
    const float* kb  = (const float*)d_in[din + 3];
    const float* vb  = (const float*)d_in[din + 5];
    const float* ob  = (const float*)d_in[din + 7];
    const float* lng = (const float*)d_in[din + 8];
    const float* lnb = (const float*)d_in[din + 9];
    const int Rq = BL * nq, Rk = BL * nk;

    gemm_ws<8, false, true, false, false><<<Rq / 128, 256, 0, stream>>>(
        Xh, Wq, qb, nullptr, Qh, HH, HH, 1);
    gemm_ws<8, false, true, false, false><<<Rk / 128, 256, 0, stream>>>(
        KeysH, Wk, kb, nullptr, Kh, HH, HH, 1);
    gemm_ws<8, false, true, true, false><<<Rk / 128, 256, 0, stream>>>(
        KeysH, Wv, vb, nullptr, Vt, HH, HH, nk);  // -> channel-major Vt
    if (nk == PPROP) {
      attn_fused<PPROP><<<dim3(nq / 16, BL * NHEADS), 256, 0, stream>>>(
          Qh, Kh, Vt, mask, Qh, nq);              // attV overwrites Qh (safe: disjoint slices)
    } else {
      attn_fused<TTOK><<<dim3(nq / 16, BL * NHEADS), 256, 0, stream>>>(
          Qh, Kh, Vt, mask, Qh, nq);
    }
    gemm_ws<8, true, false, false, false><<<Rq / 128, 256, 0, stream>>>(
        Qh, Wo, ob, Ytmp, nullptr, HH, HH, 1);
    residual_ln<<<Rq, HH, 0, stream>>>(Ytmp, Xf, Xh, lng, lnb);
  };

  // Depth loop: Bf = A2B(Bf <- A); A = B2A(A <- Bf, mask)
  for (int d = 0; d < 2; ++d) {
    run_mha(d,     Bf32, Bfh, TTOK,  Ah,  PPROP, nullptr);
    run_mha(2 + d, A32,  Ah,  PPROP, Bfh, TTOK,  qmask);
  }

  // Conv stack (1x1 convs == GEMMs over rows of A)
  gemm_ws<8, true, false, false, false><<<RA / 128, 256, 0, stream>>>(
      Ah, WH + (size_t)17 * WMAT, c1b, Ytmp, nullptr, HH, HH, 1);
  bn_stats<<<HH, 256, 0, stream>>>(Ytmp, RA, stats, stats + HH);
  bn_prelu<<<(RA * HH + 255) / 256, 256, 0, stream>>>(Ytmp, Qh, stats, stats + HH,
                                                      bn1g, bn1b, pr1, RA * HH);
  gemm_ws<8, true, false, false, false><<<RA / 128, 256, 0, stream>>>(
      Qh, WH + (size_t)18 * WMAT, c2b, Ytmp, nullptr, HH, HH, 1);
  bn_stats<<<HH, 256, 0, stream>>>(Ytmp, RA, stats, stats + HH);
  bn_prelu<<<(RA * HH + 255) / 256, 256, 0, stream>>>(Ytmp, Kh, stats, stats + HH,
                                                      bn2g, bn2b, pr2, RA * HH);
  // conv3: N padded to 16, only 4 real output channels stored -> conf
  gemm_ws<1, true, false, false, true><<<RA / 128, 256, 0, stream>>>(
      Kh, W3H, c3b, conf, nullptr, 4, 4, 1);
}
